// DecoderLayer_30399778521556
// MI455X (gfx1250) — matrix operand order repaired and checked by reading, not compiled, not run on hardware
//
#include <hip/hip_runtime.h>
#include <hip/hip_bf16.h>
#include <math.h>

// ---------------------------------------------------------------------------
// Swin-style decoder layer for MI455X (gfx1250, wave32, WMMA bf16,
// async global->LDS staging with double buffering).
// ---------------------------------------------------------------------------

typedef __attribute__((ext_vector_type(16))) __bf16 v16bf;
typedef __attribute__((ext_vector_type(8)))  float  v8f;

#define TOK    200704          // B*H*W tokens
#define CDIM   256
#define NHEAD  4
#define HDIM   64
#define NWIN   4096            // B * 32 * 32 windows
#define HID    1024
#define CHUNK  25088           // TOK / 8, multiple of 128

__device__ __forceinline__ unsigned short f32_to_bf16(float f) {
  unsigned int x = __float_as_uint(f);
  x += 0x7fffu + ((x >> 16) & 1u);       // round-to-nearest-even
  return (unsigned short)(x >> 16);
}

// Async copy of 8 bytes global -> LDS (tracked by ASYNCcnt, bypasses VGPRs).
__device__ __forceinline__ void async_b64(const void* gp, void* lds_generic) {
  unsigned lds = (unsigned)(size_t)lds_generic;            // low 32 bits = LDS addr
  unsigned long long ga = (unsigned long long)(size_t)gp;
  asm volatile("global_load_async_to_lds_b64 %0, %1, off"
               :: "v"(lds), "v"(ga) : "memory");
}
__device__ __forceinline__ void wait_async_le2() {
  asm volatile("s_wait_asynccnt 0x2" ::: "memory");
}
__device__ __forceinline__ void wait_async_0() {
  asm volatile("s_wait_asynccnt 0x0" ::: "memory");
}

union FragU { v16bf v; unsigned int u[8]; unsigned short h[16]; };

// A-fragment (16x32 bf16): lane holds row (lane&15); element e -> K =
// (lane>>4)*16 + e.  Pairs are contiguous -> dword loads.
__device__ __forceinline__ v16bf load_a_bf16(const unsigned short* A, int lda,
                                             int row0, int k0, int lane) {
  FragU f;
  const unsigned short* p =
      A + (size_t)(row0 + (lane & 15)) * lda + k0 + ((lane >> 4) << 4);
#pragma unroll
  for (int j = 0; j < 4; ++j) {
    f.u[j]     = *(const unsigned int*)(p + 2 * j);
    f.u[j + 4] = *(const unsigned int*)(p + 8 + 2 * j);
  }
  return f.v;
}

// Same fragment but converting fp32 source on the fly.
__device__ __forceinline__ v16bf load_a_f32(const float* A, int lda,
                                            int row0, int k0, int lane) {
  FragU f;
  const float* p =
      A + (size_t)(row0 + (lane & 15)) * lda + k0 + ((lane >> 4) << 4);
#pragma unroll
  for (int e = 0; e < 8; ++e) f.h[e] = f32_to_bf16(p[e]);
#pragma unroll
  for (int e = 0; e < 8; ++e) f.h[8 + e] = f32_to_bf16(p[8 + e]);
  return f.v;
}

// B-fragment (32x16 bf16): lane holds column n = (lane&15); elements e ->
// K = (lane>>4)*16 + e.  Source is column-major (Bt[n][k]) so all 16
// elements are contiguous -> 8 dword loads.
__device__ __forceinline__ v16bf load_b_cols(const unsigned short* Bt, int stride,
                                             int n0, int k0, int lane) {
  FragU f;
  const unsigned short* p =
      Bt + (size_t)(n0 + (lane & 15)) * stride + k0 + ((lane >> 4) << 4);
#pragma unroll
  for (int j = 0; j < 8; ++j) f.u[j] = *(const unsigned int*)(p + 2 * j);
  return f.v;
}

__device__ __forceinline__ v8f wmma_bf16(v16bf a, v16bf b, v8f c) {
  return __builtin_amdgcn_wmma_f32_16x16x32_bf16(false, a, false, b,
                                                 (short)0, c, false, false);
}

// ---------------------------------------------------------------------------
// LayerNorm: one wave per token (C=256 -> 8 values/lane), bf16 output
// ---------------------------------------------------------------------------
__global__ __launch_bounds__(256) void ln_k(const float* __restrict__ x,
                                            const float* __restrict__ g,
                                            const float* __restrict__ b,
                                            unsigned short* __restrict__ y) {
  const int lane = threadIdx.x & 31, wid = threadIdx.x >> 5;
  const int t = blockIdx.x * 8 + wid;
  const float* xp = x + (size_t)t * CDIM;
  float v[8];
  float s = 0.f;
#pragma unroll
  for (int i = 0; i < 8; ++i) { v[i] = xp[lane + 32 * i]; s += v[i]; }
#pragma unroll
  for (int off = 16; off > 0; off >>= 1) s += __shfl_xor(s, off, 32);
  const float mu = s * (1.f / 256.f);
  float q = 0.f;
#pragma unroll
  for (int i = 0; i < 8; ++i) { float d = v[i] - mu; q += d * d; }
#pragma unroll
  for (int off = 16; off > 0; off >>= 1) q += __shfl_xor(q, off, 32);
  const float inv = rsqrtf(q * (1.f / 256.f) + 1e-12f);
  unsigned short* yp = y + (size_t)t * CDIM;
#pragma unroll
  for (int i = 0; i < 8; ++i) {
    int c = lane + 32 * i;
    yp[c] = f32_to_bf16((v[i] - mu) * inv * g[c] + b[c]);
  }
}

// ---------------------------------------------------------------------------
// fp32 [K][N] -> bf16 transposed [N][K] weight conversion (once per launch;
// transposed layout makes GEMM B tiles layout-preserving for async staging).
// ---------------------------------------------------------------------------
__global__ __launch_bounds__(256) void cvt_t_k(const float* __restrict__ w,
                                               unsigned short* __restrict__ wt,
                                               int K, int N) {
  int i = blockIdx.x * 256 + threadIdx.x;
  if (i < K * N) {
    int k = i / N, n = i % N;
    wt[(size_t)n * K + k] = f32_to_bf16(w[i]);
  }
}

// ---------------------------------------------------------------------------
// Tiled GEMM: out[M,N] = epi(A[M,K] @ W[K,N] + bias), W given transposed
// (Wt[N][K] bf16).  128x64 macro-tile, 8 waves x 16 rows.  B tiles are
// async-copied global->LDS (double buffered, ASYNCcnt-tracked) into a
// 36-ushort-pitch column-major tile -> conflict-free contiguous fragments.
// EPI: 0 = (x+b)*scale -> bf16, 1 = x+b+resid -> f32, 2 = gelu(x+b) -> bf16.
// ---------------------------------------------------------------------------
template <int AF32, int EPI>
__global__ __launch_bounds__(256) void gemm_k(const void* __restrict__ Ap,
                                              const unsigned short* __restrict__ Wt,
                                              const float* __restrict__ bias,
                                              const float* __restrict__ resid,
                                              void* __restrict__ outp,
                                              int M, int N, int K, float scale) {
  __shared__ unsigned short Bt[2][64][36];
  const int tid = threadIdx.x, lane = tid & 31, wid = tid >> 5;
  const int m0 = blockIdx.x * 128 + wid * 16;
  const int n0 = blockIdx.y * 64;
  const int nk = K >> 5;

  v8f acc[4];
#pragma unroll
  for (int j = 0; j < 4; ++j)
#pragma unroll
    for (int r = 0; r < 8; ++r) acc[j][r] = 0.f;

  // Stage one 64(n) x 32(k) B tile into buffer bf: 512 async b64 copies.
  auto stage = [&](int bf, int k0) {
#pragma unroll
    for (int t = 0; t < 2; ++t) {
      int qi = tid + t * 256;            // 0..511
      int n  = qi >> 3;                  // 0..63
      int q4 = (qi & 7) << 2;            // 0..28 (4 ushorts per op)
      async_b64(Wt + (size_t)(n0 + n) * K + k0 + q4, &Bt[bf][n][q4]);
    }
  };

  stage(0, 0);                           // prologue: tile 0 in flight
  for (int i = 0; i < nk; ++i) {
    __syncthreads();                     // buffer (i+1)&1 free of readers
    const bool pre = (i + 1 < nk);
    if (pre) stage((i + 1) & 1, (i + 1) << 5);
    if (pre) wait_async_le2();           // retire tile i, keep i+1 in flight
    else     wait_async_0();
    __syncthreads();                     // tile i visible to all waves

    const int k0 = i << 5;
    if (pre) {                           // global_prefetch_b8 of next A slab
      const int es = AF32 ? 4 : 2;
      __builtin_prefetch(
          (const char*)Ap + ((size_t)(m0 + (lane & 15)) * K + k0 + 32) * es, 0, 1);
    }

    v16bf a;
    if (AF32) a = load_a_f32((const float*)Ap, K, m0, k0, lane);
    else      a = load_a_bf16((const unsigned short*)Ap, K, m0, k0, lane);

#pragma unroll
    for (int j = 0; j < 4; ++j) {
      v16bf bfr = load_b_cols(&Bt[i & 1][0][0], 36, j * 16, 0, lane);
      acc[j] = wmma_bf16(a, bfr, acc[j]);
    }
  }

  // C/D layout: element r -> row m0 + r + (lane>=16)*8, col n0+j*16+(lane&15)
  const int rhi = (lane >> 4) << 3;
#pragma unroll
  for (int j = 0; j < 4; ++j) {
    const int c = n0 + j * 16 + (lane & 15);
    const float bv = bias[c];
#pragma unroll
    for (int r = 0; r < 8; ++r) {
      const int row = m0 + rhi + r;
      const size_t o = (size_t)row * N + c;
      float v = acc[j][r] + bv;
      if (EPI == 0) {
        ((unsigned short*)outp)[o] = f32_to_bf16(v * scale);
      } else if (EPI == 1) {
        ((float*)outp)[o] = v + resid[o];
      } else {
        float gg = 0.5f * v * (1.f + erff(v * 0.70710678118654752f));
        ((unsigned short*)outp)[o] = f32_to_bf16(gg);
      }
    }
  }
}

// ---------------------------------------------------------------------------
// Windowed attention: one workgroup per (window, head), 4 waves.
// ---------------------------------------------------------------------------
__device__ __forceinline__ int win_token(int b, int wr, int wc, int r, int shift) {
  int i7 = r / 7, j7 = r % 7;
  int hh = wr * 7 + i7 + shift; if (hh >= 224) hh -= 224;
  int ww = wc * 7 + j7 + shift; if (ww >= 224) ww -= 224;
  return b * 50176 + hh * 224 + ww;
}
__device__ __forceinline__ int sw_label(int wr, int wc, int r) {
  int ph = wr * 7 + r / 7, pw = wc * 7 + r % 7;
  int rh = (ph < 217) ? 0 : ((ph < 221) ? 1 : 2);
  int rw = (pw < 217) ? 0 : ((pw < 221) ? 1 : 2);
  return rh * 3 + rw;
}

__global__ __launch_bounds__(128) void attn_k(const unsigned short* __restrict__ Qp,
                                              const unsigned short* __restrict__ Kp,
                                              const unsigned short* __restrict__ Vp,
                                              const unsigned char* __restrict__ bmask,
                                              const float* __restrict__ rpb,
                                              unsigned short* __restrict__ AO,
                                              int shift, int useSM) {
  __shared__ unsigned short Qs[64][68];   // Q rows (bf16), padded pitch
  __shared__ unsigned short Ks[64][68];   // K rows (bf16)
  __shared__ unsigned short Vt[64][68];   // V transposed: Vt[d][token]
  __shared__ float          Sf[64][66];   // scores f32
  __shared__ unsigned short Pb[64][68];   // softmax probs (bf16)

  const int tid = threadIdx.x, lane = tid & 31, wid = tid >> 5;
  const int h  = blockIdx.y;
  const int b  = blockIdx.x >> 10;
  const int wi = blockIdx.x & 1023;
  const int wr = wi >> 5, wc = wi & 31;

  // Async-gather Q/K rows (b64 = 4 channels per op).  Rows >= 49 are left
  // unstaged: their poison never reaches a stored value (S pads overwritten
  // with -1e30, P pad rows zeroed, O pad rows never stored).
  for (int idx = tid; idx < 64 * 16; idx += 128) {
    int r = idx >> 4, q4 = (idx & 15) << 2;
    if (r < 49) {
      size_t base = (size_t)win_token(b, wr, wc, r, shift) * CDIM + h * HDIM + q4;
      async_b64(Qp + base, &Qs[r][q4]);
      async_b64(Kp + base, &Ks[r][q4]);
    }
  }
  // V gathered transposed (manual; async cannot transpose at b16 grain).
  for (int idx = tid; idx < 64 * 64; idx += 128) {
    int r = idx >> 6, d = idx & 63;
    unsigned short v = 0;
    if (r < 49)
      v = Vp[(size_t)win_token(b, wr, wc, r, shift) * CDIM + h * HDIM + d];
    Vt[d][r] = v;
  }
  wait_async_0();
  __syncthreads();

  const int m0 = wid * 16;
  const int rhi = (lane >> 4) << 3;

  // S = Q * K^T   (M=64, N=64, K=64; B-fragment = K rows -> contiguous)
  v8f acc[4];
#pragma unroll
  for (int j = 0; j < 4; ++j)
#pragma unroll
    for (int r = 0; r < 8; ++r) acc[j][r] = 0.f;
#pragma unroll
  for (int k0 = 0; k0 < 64; k0 += 32) {
    v16bf a = load_a_bf16(&Qs[0][0], 68, m0, k0, lane);
#pragma unroll
    for (int j = 0; j < 4; ++j) {
      v16bf bf = load_b_cols(&Ks[0][0], 68, j * 16, k0, lane);
      acc[j] = wmma_bf16(a, bf, acc[j]);
    }
  }

  // Fused mask-replace (-1e9), rel-pos bias, shift mask; pad with -1e30.
#pragma unroll
  for (int j = 0; j < 4; ++j) {
    const int col = j * 16 + (lane & 15);
#pragma unroll
    for (int r = 0; r < 8; ++r) {
      const int row = m0 + rhi + r;
      float v = -1e30f;
      if (row < 49 && col < 49) {
        v = bmask[row * 49 + col] ? -1e9f : acc[j][r];
        int rel = (row / 7 - col / 7 + 6) * 13 + (row % 7 - col % 7 + 6);
        v += rpb[rel * NHEAD + h];
        if (useSM && (sw_label(wr, wc, row) != sw_label(wr, wc, col))) v -= 100.f;
      }
      Sf[row][col] = v;
    }
  }
  __syncthreads();

  // Row softmax (rows 0..48), zero padded rows of P.
  if (tid < 64) {
    if (tid < 49) {
      float mx = -1e30f;
      for (int c = 0; c < 64; ++c) mx = fmaxf(mx, Sf[tid][c]);
      float sum = 0.f;
      for (int c = 0; c < 64; ++c) {
        float e = __expf(Sf[tid][c] - mx);
        Sf[tid][c] = e;
        sum += e;
      }
      float rin = 1.f / sum;
      for (int c = 0; c < 64; ++c) Pb[tid][c] = f32_to_bf16(Sf[tid][c] * rin);
    } else {
      for (int c = 0; c < 64; ++c) Pb[tid][c] = 0;
    }
  }
  __syncthreads();

  // O = P * V   (B-fragment from transposed V -> contiguous)
  v8f oc[4];
#pragma unroll
  for (int j = 0; j < 4; ++j)
#pragma unroll
    for (int r = 0; r < 8; ++r) oc[j][r] = 0.f;
#pragma unroll
  for (int k0 = 0; k0 < 64; k0 += 32) {
    v16bf a = load_a_bf16(&Pb[0][0], 68, m0, k0, lane);
#pragma unroll
    for (int j = 0; j < 4; ++j) {
      v16bf bf = load_b_cols(&Vt[0][0], 68, j * 16, k0, lane);
      oc[j] = wmma_bf16(a, bf, oc[j]);
    }
  }

  // Scatter token-major so the projection GEMM consumes it directly.
#pragma unroll
  for (int j = 0; j < 4; ++j) {
    const int d = j * 16 + (lane & 15);
#pragma unroll
    for (int r = 0; r < 8; ++r) {
      const int row = m0 + rhi + r;
      if (row < 49)
        AO[(size_t)win_token(b, wr, wc, row, shift) * CDIM + h * HDIM + d] =
            f32_to_bf16(oc[j][r]);
    }
  }
}

// ---------------------------------------------------------------------------
// Host orchestration
// ---------------------------------------------------------------------------
extern "C" void kernel_launch(void* const* d_in, const int* in_sizes, int n_in,
                              void* d_out, int out_size, void* d_ws, size_t ws_size,
                              hipStream_t stream) {
  const float* in_x   = (const float*)d_in[0];
  const float* in_enc = (const float*)d_in[1];
  const unsigned char* bmask = (const unsigned char*)d_in[3];
  const float* ln1g = (const float*)d_in[4];
  const float* ln1b = (const float*)d_in[5];
  const float* ln2g = (const float*)d_in[6];
  const float* ln2b = (const float*)d_in[7];
  const float* ln3g = (const float*)d_in[8];
  const float* ln3b = (const float*)d_in[9];
  const float* wq1 = (const float*)d_in[10];
  const float* wk1 = (const float*)d_in[11];
  const float* wv1 = (const float*)d_in[12];
  const float* wp1 = (const float*)d_in[13];
  const float* bq1 = (const float*)d_in[14];
  const float* bk1 = (const float*)d_in[15];
  const float* bv1 = (const float*)d_in[16];
  const float* bp1 = (const float*)d_in[17];
  const float* rpb1 = (const float*)d_in[18];
  const float* wq2 = (const float*)d_in[19];
  const float* wk2 = (const float*)d_in[20];
  const float* wv2 = (const float*)d_in[21];
  const float* wp2 = (const float*)d_in[22];
  const float* bq2 = (const float*)d_in[23];
  const float* bk2 = (const float*)d_in[24];
  const float* bv2 = (const float*)d_in[25];
  const float* bp2 = (const float*)d_in[26];
  const float* rpb2 = (const float*)d_in[27];
  const float* fc1w = (const float*)d_in[28];
  const float* fc1b = (const float*)d_in[29];
  const float* fc2w = (const float*)d_in[30];
  const float* fc2b = (const float*)d_in[31];
  float* out = (float*)d_out;

  // Workspace carve-up
  const size_t TC = (size_t)TOK * CDIM;
  char* ws = (char*)d_ws;
  unsigned short* yb = (unsigned short*)ws; ws += TC * 2;   // LN output (reused 3x)
  unsigned short* qp = (unsigned short*)ws; ws += TC * 2;   // Q proj (bf16, pre-scaled)
  unsigned short* kp = (unsigned short*)ws; ws += TC * 2;   // K proj
  unsigned short* vp = (unsigned short*)ws; ws += TC * 2;   // V proj
  unsigned short* ao = (unsigned short*)ws; ws += TC * 2;   // attention output
  float* xr = (float*)ws;                   ws += TC * 4;   // running residual (f32)
  unsigned short* hch = (unsigned short*)ws; ws += (size_t)CHUNK * HID * 2;
  unsigned short* wb = (unsigned short*)ws;                 // bf16 transposed weights
  unsigned short* wq1t = wb + 0 * 65536;
  unsigned short* wk1t = wb + 1 * 65536;
  unsigned short* wv1t = wb + 2 * 65536;
  unsigned short* wp1t = wb + 3 * 65536;
  unsigned short* wq2t = wb + 4 * 65536;
  unsigned short* wk2t = wb + 5 * 65536;
  unsigned short* wv2t = wb + 6 * 65536;
  unsigned short* wp2t = wb + 7 * 65536;
  unsigned short* fc1wt = wb + 8 * 65536;
  unsigned short* fc2wt = fc1wt + 262144;

  // Weight convert + transpose ([K][N] f32 -> [N][K] bf16)
  cvt_t_k<<<256, 256, 0, stream>>>(wq1, wq1t, CDIM, CDIM);
  cvt_t_k<<<256, 256, 0, stream>>>(wk1, wk1t, CDIM, CDIM);
  cvt_t_k<<<256, 256, 0, stream>>>(wv1, wv1t, CDIM, CDIM);
  cvt_t_k<<<256, 256, 0, stream>>>(wp1, wp1t, CDIM, CDIM);
  cvt_t_k<<<256, 256, 0, stream>>>(wq2, wq2t, CDIM, CDIM);
  cvt_t_k<<<256, 256, 0, stream>>>(wk2, wk2t, CDIM, CDIM);
  cvt_t_k<<<256, 256, 0, stream>>>(wv2, wv2t, CDIM, CDIM);
  cvt_t_k<<<256, 256, 0, stream>>>(wp2, wp2t, CDIM, CDIM);
  cvt_t_k<<<1024, 256, 0, stream>>>(fc1w, fc1wt, CDIM, HID);
  cvt_t_k<<<1024, 256, 0, stream>>>(fc2w, fc2wt, HID, CDIM);

  const dim3 gC(TOK / 128, CDIM / 64);     // (1568, 4)
  const float qscale = 0.125f;             // HD^-0.5

  // ---- Block 1: plain windowed cross-attention ----
  ln_k<<<TOK / 8, 256, 0, stream>>>(in_x, ln1g, ln1b, yb);
  gemm_k<0, 0><<<gC, 256, 0, stream>>>(yb,     wq1t, bq1, nullptr, qp, TOK, CDIM, CDIM, qscale);
  gemm_k<1, 0><<<gC, 256, 0, stream>>>(in_enc, wk1t, bk1, nullptr, kp, TOK, CDIM, CDIM, 1.f);
  gemm_k<1, 0><<<gC, 256, 0, stream>>>(in_enc, wv1t, bv1, nullptr, vp, TOK, CDIM, CDIM, 1.f);
  attn_k<<<dim3(NWIN, NHEAD), 128, 0, stream>>>(qp, kp, vp, bmask, rpb1, ao, 0, 0);
  gemm_k<0, 1><<<gC, 256, 0, stream>>>(ao, wp1t, bp1, in_x, xr, TOK, CDIM, CDIM, 1.f);

  // ---- Block 2: shifted windowed cross-attention ----
  ln_k<<<TOK / 8, 256, 0, stream>>>(xr, ln2g, ln2b, yb);
  gemm_k<0, 0><<<gC, 256, 0, stream>>>(yb,     wq2t, bq2, nullptr, qp, TOK, CDIM, CDIM, qscale);
  gemm_k<1, 0><<<gC, 256, 0, stream>>>(in_enc, wk2t, bk2, nullptr, kp, TOK, CDIM, CDIM, 1.f);
  gemm_k<1, 0><<<gC, 256, 0, stream>>>(in_enc, wv2t, bv2, nullptr, vp, TOK, CDIM, CDIM, 1.f);
  attn_k<<<dim3(NWIN, NHEAD), 128, 0, stream>>>(qp, kp, vp, bmask, rpb2, ao, 3, 1);
  gemm_k<0, 1><<<gC, 256, 0, stream>>>(ao, wp2t, bp2, xr, xr, TOK, CDIM, CDIM, 1.f);

  // ---- MLP (chunked so the 1024-wide hidden stays small in scratch) ----
  ln_k<<<TOK / 8, 256, 0, stream>>>(xr, ln3g, ln3b, yb);
  for (int c = 0; c < 8; ++c) {
    const size_t off = (size_t)c * CHUNK * CDIM;
    gemm_k<0, 2><<<dim3(CHUNK / 128, HID / 64), 256, 0, stream>>>(
        yb + off, fc1wt, fc1b, nullptr, hch, CHUNK, HID, CDIM, 1.f);
    gemm_k<0, 1><<<dim3(CHUNK / 128, CDIM / 64), 256, 0, stream>>>(
        hch, fc2wt, fc2b, xr + off, out + off, CHUNK, CDIM, HID, 1.f);
  }
}
